// UnitActorCriticMultiHead_22016002359967
// MI455X (gfx1250) — matrix operand order repaired and checked by reading, not compiled, not run on hardware
//
#include <hip/hip_runtime.h>
#include <hip/hip_bf16.h>

// ---------------------------------------------------------------------------
// UnitActorCritic fused kernels for MI455X (gfx1250, wave32, WMMA + async LDS).
//
// Prep pass (once per launch, ~2.6 MB into d_ws):
//   all weights -> f16, column-major [col][k], K zero-padded (tW1: 528->544),
//   vW2 expanded to a 16-col matrix with the vector in column 0.
// Main pass (fused):
//   h1 = relu(x @ tW1 + tb1); h2 = relu(h1 @ tW2 + tb2)
//   routed head k = argmax(unit_onehot):
//     logits = relu(h2 @ pW1[k] + pb1[k]) @ pW2[k] + pb2[k]
//     value  = relu(h2 @ vW1[k] + vb1[k]) . vW2[k] + vb2[k]   (as WMMA, N==0)
//
// Workgroup g handles 64 rows n = (g&15) + 16*((g>>4)*64 + m): all rows share
// one routed head -> head GEMMs are dense. 8 waves: rowgrp = wave&3 (16 rows),
// colgrp = wave>>2 (column half). Weight chunks are double-buffered in LDS and
// staged with global_load_async_to_lds_b128 (ASYNCcnt) issued BEFORE the WMMA
// chain so the DMA overlaps compute. B fragments are loaded with inline-asm
// ds_load_b128 into distinct register groups with ONE s_wait_dscnt drain per
// chunk, so the 8 WMMAs issue back-to-back.
// ---------------------------------------------------------------------------

typedef __attribute__((ext_vector_type(16))) _Float16 v16h;
typedef __attribute__((ext_vector_type(8)))  _Float16 v8h;
typedef __attribute__((ext_vector_type(4)))  unsigned v4u;
typedef __attribute__((ext_vector_type(8)))  float    v8f;

#define N_ROWS      65536
#define OBS_DIM     512
#define NUM_TACTICS 16
#define TRUNK_H     256
#define HEAD_H      128
#define ACTION_DIM  32

#define ROWS_PER_WG 64
#define H_STRIDE    264   // 256 + 8 pad halves (bank step 4 between rows)
#define PH_STRIDE   136   // 128 + 8 pad halves
#define W_STRIDE    40    // 32 + 8 pad halves per staged weight column

// dynamic-LDS carve (units: halves)
#define WST_HALVES (256 * W_STRIDE)                 // 10240 per buffer (20 KB)
#define H_OFF      (2 * WST_HALVES)
#define H_HALVES   (ROWS_PER_WG * H_STRIDE)         // 16896
#define H2_OFF     (H_OFF + H_HALVES)
#define PH_OFF     (H2_OFF + H_HALVES)
#define PH_HALVES  (ROWS_PER_WG * PH_STRIDE)        // 8704
#define LDS_HALVES (PH_OFF + PH_HALVES)             // 62976 halves = ~123 KB

// workspace layout (halves) -- total 1,351,680 halves = ~2.6 MB
#define WS_TW1  0                         // [256][544]  (528 padded to 544)
#define WS_TW2  (WS_TW1 + 256 * 544)      // [256][256]
#define WS_PW1  (WS_TW2 + 256 * 256)      // 16 x [128][256]
#define WS_PW2  (WS_PW1 + 16 * 128 * 256) // 16 x [32][128]
#define WS_VW1  (WS_PW2 + 16 * 32 * 128)  // 16 x [128][256]
#define WS_VW2  (WS_VW1 + 16 * 128 * 256) // 16 x [16][128], col 0 = vW2
#define WS_TOTAL (WS_VW2 + 16 * 16 * 128)

// ===================== prep kernels ========================================

// src f32 row-major [b][K][C] -> dst f16 col-major [b][C][ldk], zero pad k>=K
__global__ __launch_bounds__(256)
void transpose_w_kernel(const float* __restrict__ src, _Float16* __restrict__ dst,
                        int K, int C, int ldk, int batch) {
  int total = batch * C * ldk;
  for (int i = blockIdx.x * 256 + threadIdx.x; i < total; i += gridDim.x * 256) {
    int b   = i / (C * ldk);
    int rem = i - b * C * ldk;
    int col = rem / ldk;
    int k   = rem - col * ldk;
    float v = (k < K) ? src[((size_t)b * K + k) * C + col] : 0.f;
    dst[i] = (_Float16)v;
  }
}

// vW2 (16,128,1) -> 16 x [16 cols][128 k], column 0 holds the vector
__global__ __launch_bounds__(256)
void build_vw2_kernel(const float* __restrict__ vW2, _Float16* __restrict__ dst) {
  int i = blockIdx.x * 256 + threadIdx.x;     // 16*16*128 = 32768
  if (i < 16 * 16 * 128) {
    int u = i >> 11, col = (i >> 7) & 15, k = i & 127;
    dst[i] = (_Float16)((col == 0) ? vW2[u * 128 + k] : 0.f);
  }
}

// ===================== main kernel helpers =================================

// ---- A fragment, 16x32 f16, from global f32 row (ISA 7.12.2 16-bit A layout)
__device__ __forceinline__ v16h a_frag_g32(const float* rowp, int khalf) {
  const float4* p = reinterpret_cast<const float4*>(rowp + khalf);
  float4 x0 = p[0], x1 = p[1];   // K: khalf+0..7
  float4 y0 = p[4], y1 = p[5];   // K: khalf+16..23
  v16h a;
  a[0]=(_Float16)x0.x; a[1]=(_Float16)x0.y; a[2]=(_Float16)x0.z; a[3]=(_Float16)x0.w;
  a[4]=(_Float16)x1.x; a[5]=(_Float16)x1.y; a[6]=(_Float16)x1.z; a[7]=(_Float16)x1.w;
  a[8]=(_Float16)y0.x; a[9]=(_Float16)y0.y; a[10]=(_Float16)y0.z; a[11]=(_Float16)y0.w;
  a[12]=(_Float16)y1.x; a[13]=(_Float16)y1.y; a[14]=(_Float16)y1.z; a[15]=(_Float16)y1.w;
  return a;
}

// tactic tail: only K=0..15 valid, K=16..31 zero (weights there are 0 anyway)
__device__ __forceinline__ v16h a_frag_tac(const float* rowp, int khalf) {
  const float4* p = reinterpret_cast<const float4*>(rowp + khalf);
  float4 x0 = p[0], x1 = p[1];
  v16h a;
  a[0]=(_Float16)x0.x; a[1]=(_Float16)x0.y; a[2]=(_Float16)x0.z; a[3]=(_Float16)x0.w;
  a[4]=(_Float16)x1.x; a[5]=(_Float16)x1.y; a[6]=(_Float16)x1.z; a[7]=(_Float16)x1.w;
  #pragma unroll
  for (int i = 8; i < 16; ++i) a[i] = (_Float16)0.f;
  return a;
}

// A fragment from LDS f16 row (16 B aligned -> ds_load_b128 x2)
__device__ __forceinline__ v16h a_frag_lds(const _Float16* rowp, int khalf) {
  v8h lo = *reinterpret_cast<const v8h*>(rowp + khalf);
  v8h hi = *reinterpret_cast<const v8h*>(rowp + khalf + 16);
  v16h a;
  #pragma unroll
  for (int i = 0; i < 8; ++i) { a[i] = lo[i]; a[8 + i] = hi[i]; }
  return a;
}

// Async L2->LDS copy of one 16-byte packet (gfx1250, tracked by ASYNCcnt).
__device__ __forceinline__ void async_cp_b128(const _Float16* g, _Float16* l) {
  unsigned loff = (unsigned)(size_t)l;   // generic LDS addr: low 32 bits = offset
  asm volatile("global_load_async_to_lds_b128 %0, %1, off"
               :: "v"(loff), "v"(g) : "memory");
}

__device__ __forceinline__ void wait_async0() {
  asm volatile("s_wait_asynccnt 0x0" ::: "memory");
}

// Stage a 32 x ncols pre-transposed f16 weight chunk: 4 async b128 per column.
__device__ __forceinline__ void stage_async(const _Float16* __restrict__ srcT, int ldk,
                                            int k0, _Float16* dst, int lgcols) {
  const int ncols = 1 << lgcols;
  const int total = ncols << 2;                  // 4 packets per column
  for (int i = threadIdx.x; i < total; i += 256) {
    int q   = i >> lgcols;                       // 0..3
    int col = i & (ncols - 1);
    async_cp_b128(srcT + (size_t)col * ldk + k0 + q * 8,
                  dst + col * W_STRIDE + q * 8);
  }
}

// NT-tile WMMA chain. All B fragments are pulled with inline-asm ds_load_b128
// into distinct 128-bit register groups, then a single s_wait_dscnt 0 (with
// read-write register ties so the WMMAs cannot be scheduled above it), then
// the WMMAs issue back-to-back.
template<int NT>
__device__ __forceinline__ void gemm_tiles(v8f* acc, v16h a, const _Float16* buf,
                                           int tbase, int lane) {
  const _Float16* p0 = buf + (tbase * 16 + (lane & 15)) * W_STRIDE + ((lane >> 4) << 3);
  unsigned base = (unsigned)(size_t)p0;          // LDS byte offset
  v4u lo[NT], hi[NT];
  #pragma unroll
  for (int t = 0; t < NT; ++t) {
    unsigned off = base + (unsigned)(t * 16 * W_STRIDE * sizeof(_Float16)); // +1280B/tile
    asm volatile("ds_load_b128 %0, %2\n\t"
                 "ds_load_b128 %1, %2 offset:32"
                 : "=v"(lo[t]), "=v"(hi[t])
                 : "v"(off));
  }
  if constexpr (NT == 8) {
    asm volatile("s_wait_dscnt 0x0"
                 : "+v"(lo[0]), "+v"(hi[0]), "+v"(lo[1]), "+v"(hi[1]),
                   "+v"(lo[2]), "+v"(hi[2]), "+v"(lo[3]), "+v"(hi[3]),
                   "+v"(lo[4]), "+v"(hi[4]), "+v"(lo[5]), "+v"(hi[5]),
                   "+v"(lo[6]), "+v"(hi[6]), "+v"(lo[7]), "+v"(hi[7]));
  } else if constexpr (NT == 4) {
    asm volatile("s_wait_dscnt 0x0"
                 : "+v"(lo[0]), "+v"(hi[0]), "+v"(lo[1]), "+v"(hi[1]),
                   "+v"(lo[2]), "+v"(hi[2]), "+v"(lo[3]), "+v"(hi[3]));
  } else {
    asm volatile("s_wait_dscnt 0x0" : "+v"(lo[0]), "+v"(hi[0]));
  }
  #pragma unroll
  for (int t = 0; t < NT; ++t) {
    v8h l8 = __builtin_bit_cast(v8h, lo[t]);
    v8h h8 = __builtin_bit_cast(v8h, hi[t]);
    v16h b;
    #pragma unroll
    for (int i = 0; i < 8; ++i) { b[i] = l8[i]; b[8 + i] = h8[i]; }
    acc[t] = __builtin_amdgcn_wmma_f32_16x16x32_f16(false, a, false, b,
                                                    (short)0, acc[t], false, false);
  }
}

// ===================== main fused kernel ===================================

__global__ __launch_bounds__(256)
void uac_fused_kernel(const float* __restrict__ obs,
                      const float* __restrict__ tac,
                      const float* __restrict__ uonehot,
                      const _Float16* __restrict__ wsT,
                      const float* __restrict__ tb1, const float* __restrict__ tb2,
                      const float* __restrict__ pb1, const float* __restrict__ pb2,
                      const float* __restrict__ vb1, const float* __restrict__ vb2,
                      float* __restrict__ out_logits, float* __restrict__ out_values) {
  extern __shared__ _Float16 lds[];
  _Float16* H  = lds + H_OFF;    // h1, then A-source for L2
  _Float16* H2 = lds + H2_OFF;   // h2, A-source for both heads
  _Float16* PH = lds + PH_OFF;   // ph, then vh
  auto Wbuf = [&](int i) { return lds + i * WST_HALVES; };

  const int tid    = threadIdx.x;
  const int lane   = tid & 31;
  const int wave   = tid >> 5;
  const int rowgrp = wave & 3;         // 4 row groups of 16 rows
  const int colgrp = wave >> 2;        // 2 column halves
  const int lcol   = lane & 15;        // N within tile; also A-frag row M
  const int khalf  = (lane >> 4) << 3; // A-frag K sub-offset (0 or 8)
  const int lhalf  = lane >> 4;        // C layout: M = r + 8*lhalf

  const int wg    = blockIdx.x;
  const int unit  = wg & 15;           // residue class of rows handled here
  const int jbase = (wg >> 4) * ROWS_PER_WG;
  const int row0  = rowgrp * 16;

  // hard routing: argmax of unit_ids_onehot at first row of this workgroup
  int uu = 0;
  {
    const float* uo = uonehot + (size_t)(unit + 16 * jbase) * 16;
    float best = uo[0];
    #pragma unroll
    for (int i = 1; i < 16; ++i) { float v = uo[i]; if (v > best) { best = v; uu = i; } }
  }

  const long grow     = (long)unit + 16L * (jbase + row0 + lcol); // lane's A row
  const float* obsrow = obs + grow * OBS_DIM;
  const float* tacrow = tac + grow * NUM_TACTICS;

  const _Float16* T1  = wsT + WS_TW1;                    // ldk 544
  const _Float16* T2  = wsT + WS_TW2;                    // ldk 256
  const _Float16* P1  = wsT + WS_PW1 + uu * 128 * 256;   // ldk 256
  const _Float16* P2  = wsT + WS_PW2 + uu * 32 * 128;    // ldk 128
  const _Float16* V1  = wsT + WS_VW1 + uu * 128 * 256;   // ldk 256
  const _Float16* V2  = wsT + WS_VW2 + uu * 16 * 128;    // ldk 128

  v8f acc[8];

  // ---------------- Layer 1: 544(padded) -> 256, 17 uniform K=32 chunks ----
  #pragma unroll
  for (int t = 0; t < 8; ++t)
    #pragma unroll
    for (int i = 0; i < 8; ++i) acc[t][i] = 0.f;

  stage_async(T1, 544, 0, Wbuf(0), 8);
  wait_async0();
  __syncthreads();
  for (int c = 0; c < 17; ++c) {
    if (c < 16) stage_async(T1, 544, (c + 1) * 32, Wbuf((c + 1) & 1), 8); // DMA under WMMAs
    if (c < 15) __builtin_prefetch(obsrow + (c + 1) * 32, 0, 0);
    v16h a = (c < 16) ? a_frag_g32(obsrow + c * 32, khalf) : a_frag_tac(tacrow, khalf);
    gemm_tiles<8>(acc, a, Wbuf(c & 1), colgrp * 8, lane);
    wait_async0();
    __syncthreads();
  }
  #pragma unroll
  for (int t = 0; t < 8; ++t) {
    int col = colgrp * 128 + t * 16 + lcol;
    float bia = tb1[col];
    #pragma unroll
    for (int r = 0; r < 8; ++r) {
      float v = acc[t][r] + bia;
      H[(row0 + r + 8 * lhalf) * H_STRIDE + col] = (_Float16)(v > 0.f ? v : 0.f);
    }
  }

  // ---------------- Layer 2: 256 -> 256 ------------------------------------
  #pragma unroll
  for (int t = 0; t < 8; ++t)
    #pragma unroll
    for (int i = 0; i < 8; ++i) acc[t][i] = 0.f;

  stage_async(T2, 256, 0, Wbuf(0), 8);
  wait_async0();
  __syncthreads();                        // also fences h1 stores
  for (int c = 0; c < 8; ++c) {
    if (c < 7) stage_async(T2, 256, (c + 1) * 32, Wbuf((c + 1) & 1), 8);
    v16h a = a_frag_lds(H + (row0 + lcol) * H_STRIDE + c * 32, khalf);
    gemm_tiles<8>(acc, a, Wbuf(c & 1), colgrp * 8, lane);
    wait_async0();
    __syncthreads();
  }
  #pragma unroll
  for (int t = 0; t < 8; ++t) {
    int col = colgrp * 128 + t * 16 + lcol;
    float bia = tb2[col];
    #pragma unroll
    for (int r = 0; r < 8; ++r) {
      float v = acc[t][r] + bia;
      H2[(row0 + r + 8 * lhalf) * H_STRIDE + col] = (_Float16)(v > 0.f ? v : 0.f);
    }
  }

  // ---------------- Policy head 1: 256 -> 128 (routed) ---------------------
  v8f hacc[4];
  #pragma unroll
  for (int t = 0; t < 4; ++t)
    #pragma unroll
    for (int i = 0; i < 8; ++i) hacc[t][i] = 0.f;

  stage_async(P1, 256, 0, Wbuf(0), 7);
  wait_async0();
  __syncthreads();                        // also fences h2 stores
  for (int c = 0; c < 8; ++c) {
    if (c < 7) stage_async(P1, 256, (c + 1) * 32, Wbuf((c + 1) & 1), 7);
    v16h a = a_frag_lds(H2 + (row0 + lcol) * H_STRIDE + c * 32, khalf);
    gemm_tiles<4>(hacc, a, Wbuf(c & 1), colgrp * 4, lane);
    wait_async0();
    __syncthreads();
  }
  #pragma unroll
  for (int t = 0; t < 4; ++t) {
    int col = colgrp * 64 + t * 16 + lcol;
    float bia = pb1[uu * HEAD_H + col];
    #pragma unroll
    for (int r = 0; r < 8; ++r) {
      float v = hacc[t][r] + bia;
      PH[(row0 + r + 8 * lhalf) * PH_STRIDE + col] = (_Float16)(v > 0.f ? v : 0.f);
    }
  }

  // ---------------- Policy head 2: 128 -> 32 -------------------------------
  v8f lacc;
  #pragma unroll
  for (int i = 0; i < 8; ++i) lacc[i] = 0.f;

  stage_async(P2, 128, 0, Wbuf(0), 5);
  wait_async0();
  __syncthreads();                        // also fences ph stores
  for (int c = 0; c < 4; ++c) {
    if (c < 3) stage_async(P2, 128, (c + 1) * 32, Wbuf((c + 1) & 1), 5);
    v16h a = a_frag_lds(PH + (row0 + lcol) * PH_STRIDE + c * 32, khalf);
    gemm_tiles<1>(&lacc, a, Wbuf(c & 1), colgrp, lane);  // wave owns one 16-col tile
    wait_async0();
    __syncthreads();
  }
  {
    int colb = colgrp * 16 + lcol;
    float bia = pb2[uu * ACTION_DIM + colb];
    #pragma unroll
    for (int r = 0; r < 8; ++r) {
      long n = (long)unit + 16L * (jbase + row0 + r + 8 * lhalf);
      out_logits[n * ACTION_DIM + colb] = lacc[r] + bia;
    }
  }

  // ---------------- Value head 1: 256 -> 128 (routed), vh overwrites PH ----
  #pragma unroll
  for (int t = 0; t < 4; ++t)
    #pragma unroll
    for (int i = 0; i < 8; ++i) hacc[t][i] = 0.f;

  stage_async(V1, 256, 0, Wbuf(0), 7);
  wait_async0();
  __syncthreads();
  for (int c = 0; c < 8; ++c) {
    if (c < 7) stage_async(V1, 256, (c + 1) * 32, Wbuf((c + 1) & 1), 7);
    v16h a = a_frag_lds(H2 + (row0 + lcol) * H_STRIDE + c * 32, khalf);
    gemm_tiles<4>(hacc, a, Wbuf(c & 1), colgrp * 4, lane);
    wait_async0();
    __syncthreads();
  }
  #pragma unroll
  for (int t = 0; t < 4; ++t) {
    int col = colgrp * 64 + t * 16 + lcol;
    float bia = vb1[uu * HEAD_H + col];
    #pragma unroll
    for (int r = 0; r < 8; ++r) {
      float v = hacc[t][r] + bia;
      PH[(row0 + r + 8 * lhalf) * PH_STRIDE + col] = (_Float16)(v > 0.f ? v : 0.f);
    }
  }

  // ---------------- Value head 2: 128 -> 1 as WMMA (vW2 in column 0) -------
  v8f vacc;
  #pragma unroll
  for (int i = 0; i < 8; ++i) vacc[i] = 0.f;

  stage_async(V2, 128, 0, Wbuf(0), 4);
  wait_async0();
  __syncthreads();                        // also fences vh stores
  for (int c = 0; c < 4; ++c) {
    if (c < 3) stage_async(V2, 128, (c + 1) * 32, Wbuf((c + 1) & 1), 4);
    v16h a = a_frag_lds(PH + (row0 + lcol) * PH_STRIDE + c * 32, khalf);
    gemm_tiles<1>(&vacc, a, Wbuf(c & 1), 0, lane);   // result lives in N==0 lanes
    wait_async0();
    __syncthreads();
  }
  if (colgrp == 0 && lcol == 0) {         // lanes 0 & 16 of rowgrp waves
    float bia = vb2[uu];
    #pragma unroll
    for (int r = 0; r < 8; ++r) {
      long n = (long)unit + 16L * (jbase + row0 + r + 8 * lhalf);
      out_values[n] = vacc[r] + bia;
    }
  }
}

extern "C" void kernel_launch(void* const* d_in, const int* in_sizes, int n_in,
                              void* d_out, int out_size, void* d_ws, size_t ws_size,
                              hipStream_t stream) {
  const float* obs = (const float*)d_in[0];
  const float* tac = (const float*)d_in[1];
  const float* uoh = (const float*)d_in[2];
  const float* tW1 = (const float*)d_in[3];
  const float* tb1 = (const float*)d_in[4];
  const float* tW2 = (const float*)d_in[5];
  const float* tb2 = (const float*)d_in[6];
  const float* pW1 = (const float*)d_in[7];
  const float* pb1 = (const float*)d_in[8];
  const float* pW2 = (const float*)d_in[9];
  const float* pb2 = (const float*)d_in[10];
  const float* vW1 = (const float*)d_in[11];
  const float* vb1 = (const float*)d_in[12];
  const float* vW2 = (const float*)d_in[13];
  const float* vb2 = (const float*)d_in[14];
  float* out_logits = (float*)d_out;
  float* out_values = out_logits + (size_t)N_ROWS * ACTION_DIM;

  _Float16* ws = (_Float16*)d_ws;   // needs WS_TOTAL*2 ~ 2.6 MB of scratch

  // prep: f16 column-major (K-padded) weight images
  transpose_w_kernel<<<dim3(544), dim3(256), 0, stream>>>(tW1, ws + WS_TW1, 528, 256, 544, 1);
  transpose_w_kernel<<<dim3(256), dim3(256), 0, stream>>>(tW2, ws + WS_TW2, 256, 256, 256, 1);
  transpose_w_kernel<<<dim3(2048), dim3(256), 0, stream>>>(pW1, ws + WS_PW1, 256, 128, 256, 16);
  transpose_w_kernel<<<dim3(256), dim3(256), 0, stream>>>(pW2, ws + WS_PW2, 128, 32, 128, 16);
  transpose_w_kernel<<<dim3(2048), dim3(256), 0, stream>>>(vW1, ws + WS_VW1, 256, 128, 256, 16);
  build_vw2_kernel<<<dim3(128), dim3(256), 0, stream>>>(vW2, ws + WS_VW2);

  const size_t ldsBytes = (size_t)LDS_HALVES * sizeof(_Float16);  // ~123 KB (<320 KB/WGP)
  (void)hipFuncSetAttribute(reinterpret_cast<const void*>(uac_fused_kernel),
                            hipFuncAttributeMaxDynamicSharedMemorySize, (int)ldsBytes);

  uac_fused_kernel<<<dim3(N_ROWS / ROWS_PER_WG), dim3(256), ldsBytes, stream>>>(
      obs, tac, uoh, ws, tb1, tb2, pb1, pb2, vb1, vb2, out_logits, out_values);
}
